// VQVAE_16114717295071
// MI455X (gfx1250) — compile-verified, matrix operand order checked by reading, not run on hardware
//
#include <hip/hip_runtime.h>
#include <hip/hip_bf16.h>

// ---------------------------------------------------------------------------
// CDNA5 (gfx1250) VQ-VAE forward: all GEMM-shaped work runs on
// v_wmma_f32_16x16x32_bf16 (wave32, D = A(16x32) * B(32x16) + C(16x16 f32)).
// Conv: 16-cout x 64-pixel tiles per wave (4 WMMA / k-step, one A fragment).
//       All spatial dims are square powers of two -> pure 32-bit shift/mask
//       addressing, no 64-bit div/mod anywhere in device code.
//       B-tile fill moves raw bf16 bits (no convert round trip).
// VQ:   z-row A fragments held in 128 VGPRs across the whole codebook scan.
// ---------------------------------------------------------------------------

typedef __attribute__((ext_vector_type(16))) __bf16 v16bf;
typedef __attribute__((ext_vector_type(8)))  __bf16 v8bf;
typedef __attribute__((ext_vector_type(8)))  float  v8f;

__device__ __forceinline__ unsigned short f2bf(float f) {
  unsigned int u = __float_as_uint(f);
  u += 0x7FFFu + ((u >> 16) & 1u);        // round-to-nearest-even
  return (unsigned short)(u >> 16);
}
__device__ __forceinline__ float bf2f(unsigned short b) {
  return __uint_as_float(((unsigned int)b) << 16);
}

// --------------------------- elementwise helpers ---------------------------

__global__ void k_f32_to_bf16(const float* __restrict__ src,
                              unsigned short* __restrict__ dst, int n) {
  int i = blockIdx.x * blockDim.x + threadIdx.x;
  if (i < n) dst[i] = f2bf(src[i]);
}

// Pack weights into A-matrix layout: [Cout][Kpad] bf16, K = Cin*9, zero pad.
// conv  (OIHW):   A[o][k]            = w[o*K + k]
// deconv (IOHW):  A[o][ci*9 + r]     = w[(ci*Cout + o)*9 + r]
__global__ void k_pack_w(const float* __restrict__ w,
                         unsigned short* __restrict__ dst,
                         int Cout, int Cin, int deconv, int Kpad) {
  int i = blockIdx.x * blockDim.x + threadIdx.x;
  int total = Cout * Kpad;
  if (i >= total) return;
  int o = i / Kpad;
  int k = i - o * Kpad;
  int K = Cin * 9;
  float v = 0.f;
  if (k < K) {
    if (!deconv) {
      v = w[o * K + k];
    } else {
      int ci = k / 9, r = k - ci * 9;
      v = w[(ci * Cout + o) * 9 + r];
    }
  }
  dst[i] = f2bf(v);
}

__global__ void k_norms_f32(const float* __restrict__ rows,
                            float* __restrict__ out, int n, int K) {
  int i = blockIdx.x * blockDim.x + threadIdx.x;
  if (i >= n) return;
  const float* p = rows + (long)i * K;
  float s = 0.f;
  for (int k = 0; k < K; ++k) { float v = p[k]; s += v * v; }
  out[i] = s;
}

__global__ void k_norms_bf16(const unsigned short* __restrict__ rows,
                             float* __restrict__ out, int n, int K) {
  int i = blockIdx.x * blockDim.x + threadIdx.x;
  if (i >= n) return;
  const unsigned short* p = rows + (long)i * K;
  float s = 0.f;
  for (int k = 0; k < K; ++k) { float v = bf2f(p[k]); s += v * v; }
  out[i] = s;
}

// z (bf16, [16][512][32][32]) -> zf (bf16, [16384][512] channels-last rows)
__global__ void k_transpose_z(const unsigned short* __restrict__ z,
                              unsigned short* __restrict__ zf, int total) {
  int i = blockIdx.x * blockDim.x + threadIdx.x;
  if (i >= total) return;
  int c   = i & 511;
  int row = i >> 9;
  int n   = row >> 10;
  int hw  = row & 1023;
  zf[i] = z[(((n << 9) + c) << 10) + hw];
}

// zq (bf16, [16][512][32][32]) = codebook_f32[indices[row]][c]
__global__ void k_gather_zq(const float* __restrict__ cbf,
                            const int* __restrict__ ind,
                            unsigned short* __restrict__ zq, int total) {
  int i = blockIdx.x * blockDim.x + threadIdx.x;
  if (i >= total) return;
  int hw = i & 1023;
  int t  = i >> 10;
  int c  = t & 511;
  int n  = t >> 9;
  int row = (n << 10) + hw;
  zq[i] = f2bf(cbf[((long)ind[row] << 9) + c]);
}

__global__ void k_copy_idx(const int* __restrict__ src,
                           int* __restrict__ dst, int n) {
  int i = blockIdx.x * blockDim.x + threadIdx.x;
  if (i < n) dst[i] = src[i];
}

// ------------------------- implicit-GEMM conv/deconv -----------------------
// Block = 256 threads = 8 waves. Wave w computes output tile:
//   cout in [blockIdx.y*128 + w*16, +16), pixels in [blockIdx.x*64, +64).
// Spatial dims: Hin=Win=1<<ishift, Hout=Wout=1<<oshift (all square pow2).
// K loop in steps of 32 with A = packed weights [Cout][Kpad] (bf16, global),
// B = im2col tile staged in LDS as [pixel][k] (64x32 bf16 = 4 KB), raw bits.
// Each k-step: 1 A fragment, 4 WMMAs (one per 16-pixel group).
// Fragment layouts per ISA 7.12.2:
//   A: lane holds row m=lane%16, K in {8*(lane/16)+0..7} u {16+8*(lane/16)+0..7}
//   B: mirrored (col n=lane%16)
//   C/D: vgpr r -> row m=r+8*(lane/16), col n=lane%16
__global__ __launch_bounds__(256)
void k_conv_wmma(const unsigned short* __restrict__ in,
                 const unsigned short* __restrict__ wA,
                 const float* __restrict__ bias,
                 unsigned short* __restrict__ out_bf,
                 float* __restrict__ out_f32,
                 int N, int Cin, int ishift, int Cout, int oshift,
                 int stride, int deconv, int act, int Kpad) {
  __shared__ unsigned short Btile[64 * 32];

  const int tid  = threadIdx.x;
  const int wave = tid >> 5;
  const int lane = tid & 31;
  const int half = lane >> 4;
  const int ln16 = lane & 15;

  const int Win  = 1 << ishift;             // == Hin
  const int Wout = 1 << oshift;             // == Hout
  const int omask = Wout - 1;

  const int npix     = N << (2 * oshift);
  const int pixbase  = blockIdx.x * 64;
  const int coutbase = blockIdx.y * 128 + wave * 16;

  int arow = coutbase + ln16;
  if (arow >= Cout) arow = Cout - 1;          // keep EXEC full; skip at store
  const unsigned short* aprow = wA + (long)arow * Kpad;

  // This thread fills the same 8 pixels of the B tile every k-step:
  // hoist pixel decomposition out of the K loop (shift/mask only).
  int  fill_base[8];                          // (n*Cin) << 2*ishift
  int  fill_oh[8], fill_ow[8];
  const int kk = tid & 31;                    // k-lane within the 32-wide tile
  #pragma unroll
  for (int e = 0; e < 8; ++e) {
    int p = (tid + e * 256) >> 5;             // pixel slot 0..63
    int pix = pixbase + p;
    if (pix >= npix) pix = npix - 1;          // clamp; tiles are exact anyway
    int ow = pix & omask;
    int oh = (pix >> oshift) & omask;
    int n  = pix >> (2 * oshift);
    fill_base[e] = (n * Cin) << (2 * ishift);
    fill_oh[e] = oh;
    fill_ow[e] = ow;
  }
  // Incremental k -> (c, r=kh*3+kw) decomposition: k = k0 + kk, k += 32 each
  // iteration  =>  c += 3, r += 5 (carry at 9). No div/mod in the hot loop.
  int cc = kk / 9;
  int rr = kk - cc * 9;

  v8f acc[4] = {};

  for (int k0 = 0; k0 < Kpad; k0 += 32) {
    __syncthreads();
    {
      int kh = rr / 3;
      int kw = rr - kh * 3;
      #pragma unroll
      for (int e = 0; e < 8; ++e) {
        int p = (tid + e * 256) >> 5;
        unsigned short v = 0;                 // raw bf16 bits, no converts
        if (cc < Cin) {                       // cc<Cin  <=>  k<K
          int oh = fill_oh[e], ow = fill_ow[e];
          int ih, iw;
          bool ok;
          if (!deconv) {                      // pad=1 conv, stride 1 or 2
            ih = oh * stride + kh - 1;
            iw = ow * stride + kw - 1;
            ok = (ih >= 0) & (ih < Win) & (iw >= 0) & (iw < Win);
          } else {                            // ConvTranspose2d k3 s2 p1 op1
            int th = oh + 1 - kh;
            int tw = ow + 1 - kw;
            ih = th >> 1; iw = tw >> 1;
            ok = (th >= 0) & (tw >= 0) & ((th & 1) == 0) & ((tw & 1) == 0)
               & (ih < Win) & (iw < Win);
          }
          if (ok) {
            int idx = fill_base[e] + (cc << (2 * ishift)) + (ih << ishift) + iw;
            v = in[idx];
          }
        }
        Btile[(p << 5) + kk] = v;
      }
      cc += 3; rr += 5;
      if (rr >= 9) { rr -= 9; ++cc; }
    }
    __syncthreads();

    const unsigned short* ap = aprow + k0 + 8 * half;
    v8bf alo = *(const v8bf*)ap;
    v8bf ahi = *(const v8bf*)(ap + 16);
    v16bf a;
    #pragma unroll
    for (int e2 = 0; e2 < 8; ++e2) { a[e2] = alo[e2]; a[e2 + 8] = ahi[e2]; }

    #pragma unroll
    for (int g = 0; g < 4; ++g) {
      const unsigned short* bp = &Btile[((g * 16 + ln16) << 5) + 8 * half];
      v8bf blo = *(const v8bf*)bp;
      v8bf bhi = *(const v8bf*)(bp + 16);
      v16bf b;
      #pragma unroll
      for (int e2 = 0; e2 < 8; ++e2) { b[e2] = blo[e2]; b[e2 + 8] = bhi[e2]; }
      acc[g] = __builtin_amdgcn_wmma_f32_16x16x32_bf16(false, a, false, b,
                                                       (short)0, acc[g],
                                                       false, false);
    }
  }

  #pragma unroll
  for (int g = 0; g < 4; ++g) {
    int pix = pixbase + g * 16 + ln16;
    if (pix < npix) {
      int ow = pix & omask;
      int oh = (pix >> oshift) & omask;
      int n  = pix >> (2 * oshift);
      #pragma unroll
      for (int r = 0; r < 8; ++r) {
        int co = coutbase + r + 8 * half;
        if (co < Cout) {
          float v = acc[g][r] + bias[co];
          if (act == 1)      v = v > 0.f ? v : 0.f;            // relu
          else if (act == 2) v = 1.f / (1.f + __expf(-v));     // sigmoid
          int oidx = ((n * Cout + co) << (2 * oshift)) + (oh << oshift) + ow;
          if (out_bf) out_bf[oidx] = f2bf(v);
          else        out_f32[oidx] = v;
        }
      }
    }
  }
}

// ------------------------------- VQ argmin ---------------------------------
// d2 = |z|^2 - 2 z.c + |c|^2 ; dot via WMMA (16 rows x 16 codes, K=512).
// The lane's z-row A fragments (16 k-steps, 128 VGPRs) are loaded ONCE and
// reused across all 512 codebook tiles; only the bf16 codebook (8.4 MB,
// L2-resident) streams in. 16 back-to-back v_wmma per code tile.
__global__ __launch_bounds__(256)
void k_vq_argmin(const unsigned short* __restrict__ zf,
                 const unsigned short* __restrict__ cb,
                 const float* __restrict__ z2,
                 const float* __restrict__ c2,
                 int* __restrict__ indices,
                 int ncodes, int K) {
  __shared__ float svals[8][16][16];
  __shared__ int   sidxs[8][16][16];

  const int tid  = threadIdx.x;
  const int wave = tid >> 5;
  const int lane = tid & 31;
  const int half = lane >> 4;
  const int ln16 = lane & 15;
  const int rowbase = (blockIdx.x * 8 + wave) * 16;

  // Hoist all A fragments for this lane's z-row into registers.
  const unsigned short* aprow = zf + ((long)(rowbase + ln16) * K);
  v16bf areg[16];
  #pragma unroll
  for (int k0 = 0; k0 < 16; ++k0) {
    const unsigned short* ap = aprow + k0 * 32 + 8 * half;
    v8bf alo = *(const v8bf*)ap;
    v8bf ahi = *(const v8bf*)(ap + 16);
    #pragma unroll
    for (int e = 0; e < 8; ++e) { areg[k0][e] = alo[e]; areg[k0][e + 8] = ahi[e]; }
  }

  float best[8]; int bidx[8];
  #pragma unroll
  for (int r = 0; r < 8; ++r) { best[r] = 3.4e38f; bidx[r] = 0; }

  for (int c0 = 0; c0 < ncodes; c0 += 16) {
    const unsigned short* bprow = cb + ((long)(c0 + ln16) * K);
    // prefetch next code tile's row (global_prefetch_b8) while computing
    int pfrow = c0 + 16 + ln16;
    if (pfrow >= ncodes) pfrow = ncodes - 1;
    __builtin_prefetch(cb + (long)pfrow * K, 0, 0);

    v8f acc = {};
    #pragma unroll
    for (int k0 = 0; k0 < 16; ++k0) {
      const unsigned short* bp = bprow + k0 * 32 + 8 * half;
      v8bf blo = *(const v8bf*)bp;
      v8bf bhi = *(const v8bf*)(bp + 16);
      v16bf b;
      #pragma unroll
      for (int e = 0; e < 8; ++e) { b[e] = blo[e]; b[e + 8] = bhi[e]; }
      acc = __builtin_amdgcn_wmma_f32_16x16x32_bf16(false, areg[k0], false, b,
                                                    (short)0, acc, false, false);
    }
    int code = c0 + ln16;
    float cc = c2[code];
    #pragma unroll
    for (int r = 0; r < 8; ++r) {
      int row = rowbase + r + 8 * half;
      float d = z2[row] - 2.f * acc[r] + cc;
      if (d < best[r]) { best[r] = d; bidx[r] = code; }
    }
  }

  #pragma unroll
  for (int r = 0; r < 8; ++r) {
    svals[wave][r + 8 * half][ln16] = best[r];
    sidxs[wave][r + 8 * half][ln16] = bidx[r];
  }
  __syncthreads();
  if (lane < 16) {
    float bv = svals[wave][lane][0];
    int   bi = sidxs[wave][lane][0];
    #pragma unroll
    for (int n2 = 1; n2 < 16; ++n2) {
      float v  = svals[wave][lane][n2];
      int   i2 = sidxs[wave][lane][n2];
      if (v < bv || (v == bv && i2 < bi)) { bv = v; bi = i2; }
    }
    indices[rowbase + lane] = bi;
  }
}

// ------------------------------- launcher ----------------------------------

extern "C" void kernel_launch(void* const* d_in, const int* in_sizes, int n_in,
                              void* d_out, int out_size, void* d_ws, size_t ws_size,
                              hipStream_t stream) {
  (void)in_sizes; (void)n_in; (void)out_size; (void)ws_size;

  const float* x   = (const float*)d_in[0];
  const float* w1  = (const float*)d_in[1];
  const float* b1  = (const float*)d_in[2];
  const float* w2  = (const float*)d_in[3];
  const float* b2  = (const float*)d_in[4];
  const float* w3  = (const float*)d_in[5];
  const float* b3  = (const float*)d_in[6];
  const float* w4  = (const float*)d_in[7];
  const float* b4  = (const float*)d_in[8];
  const float* cbf = (const float*)d_in[9];
  const float* d1w = (const float*)d_in[10];
  const float* d1b = (const float*)d_in[11];
  const float* d2w = (const float*)d_in[12];
  const float* d2b = (const float*)d_in[13];
  const float* d3w = (const float*)d_in[14];
  const float* d3b = (const float*)d_in[15];
  const float* wo  = (const float*)d_in[16];
  const float* bo  = (const float*)d_in[17];

  const int Nb = 16;
  char* ws = (char*)d_ws;
  size_t off = 0;
  auto carve = [&](size_t bytes) -> void* {
    void* p = ws + off;
    off = (off + bytes + 255) & ~(size_t)255;
    return p;
  };

  // activations (bf16)
  unsigned short* x_bf = (unsigned short*)carve((size_t)Nb * 3   * 256 * 256 * 2);
  unsigned short* h1   = (unsigned short*)carve((size_t)Nb * 64  * 256 * 256 * 2);
  unsigned short* h2   = (unsigned short*)carve((size_t)Nb * 128 * 128 * 128 * 2);
  unsigned short* h3   = (unsigned short*)carve((size_t)Nb * 256 * 64  * 64  * 2);
  unsigned short* zt   = (unsigned short*)carve((size_t)Nb * 512 * 32  * 32  * 2);
  unsigned short* zf   = (unsigned short*)carve((size_t)16384 * 512 * 2);
  unsigned short* zq   = (unsigned short*)carve((size_t)Nb * 512 * 32  * 32  * 2);
  unsigned short* g1   = (unsigned short*)carve((size_t)Nb * 256 * 64  * 64  * 2);
  unsigned short* g2   = (unsigned short*)carve((size_t)Nb * 128 * 128 * 128 * 2);
  unsigned short* g3   = (unsigned short*)carve((size_t)Nb * 64  * 256 * 256 * 2);
  unsigned short* cb_bf= (unsigned short*)carve((size_t)8192 * 512 * 2);
  // packed weights [Cout][Kpad] bf16
  unsigned short* w1p  = (unsigned short*)carve((size_t)64  * 32   * 2);
  unsigned short* w2p  = (unsigned short*)carve((size_t)128 * 576  * 2);
  unsigned short* w3p  = (unsigned short*)carve((size_t)256 * 1152 * 2);
  unsigned short* w4p  = (unsigned short*)carve((size_t)512 * 2304 * 2);
  unsigned short* d1p  = (unsigned short*)carve((size_t)256 * 4608 * 2);
  unsigned short* d2p  = (unsigned short*)carve((size_t)128 * 2304 * 2);
  unsigned short* d3p  = (unsigned short*)carve((size_t)64  * 1152 * 2);
  unsigned short* wop  = (unsigned short*)carve((size_t)3   * 576  * 2);
  float* z2  = (float*)carve((size_t)16384 * 4);
  float* c2  = (float*)carve((size_t)8192 * 4);
  int*   ind = (int*)  carve((size_t)16384 * 4);

  auto grid1 = [](long n) { return dim3((unsigned)((n + 255) / 256)); };

  // ---- conversions & packing ----
  k_f32_to_bf16<<<grid1((long)Nb * 3 * 256 * 256), 256, 0, stream>>>(
      x, x_bf, Nb * 3 * 256 * 256);
  k_f32_to_bf16<<<grid1((long)8192 * 512), 256, 0, stream>>>(
      cbf, cb_bf, 8192 * 512);

  k_pack_w<<<grid1((long)64  * 32  ), 256, 0, stream>>>(w1,  w1p, 64,  3,   0, 32);
  k_pack_w<<<grid1((long)128 * 576 ), 256, 0, stream>>>(w2,  w2p, 128, 64,  0, 576);
  k_pack_w<<<grid1((long)256 * 1152), 256, 0, stream>>>(w3,  w3p, 256, 128, 0, 1152);
  k_pack_w<<<grid1((long)512 * 2304), 256, 0, stream>>>(w4,  w4p, 512, 256, 0, 2304);
  k_pack_w<<<grid1((long)256 * 4608), 256, 0, stream>>>(d1w, d1p, 256, 512, 1, 4608);
  k_pack_w<<<grid1((long)128 * 2304), 256, 0, stream>>>(d2w, d2p, 128, 256, 1, 2304);
  k_pack_w<<<grid1((long)64  * 1152), 256, 0, stream>>>(d3w, d3p, 64,  128, 1, 1152);
  k_pack_w<<<grid1((long)3   * 576 ), 256, 0, stream>>>(wo,  wop, 3,   64,  0, 576);

  k_norms_f32<<<grid1(8192), 256, 0, stream>>>(cbf, c2, 8192, 512);

  auto conv = [&](const unsigned short* in, const unsigned short* wA,
                  const float* bias, unsigned short* obf, float* of32,
                  int Cin, int ishift, int Cout, int oshift,
                  int stride, int deconv, int act, int Kpad) {
    long npix = (long)Nb << (2 * oshift);
    dim3 grid((unsigned)((npix + 63) / 64), (unsigned)((Cout + 127) / 128));
    k_conv_wmma<<<grid, 256, 0, stream>>>(in, wA, bias, obf, of32,
        Nb, Cin, ishift, Cout, oshift, stride, deconv, act, Kpad);
  };

  // ---- encoder ----  (ishift/oshift = log2 of square spatial dims)
  conv(x_bf, w1p, b1, h1, nullptr, 3,   8, 64,  8, 1, 0, 1, 32);
  conv(h1,   w2p, b2, h2, nullptr, 64,  8, 128, 7, 2, 0, 1, 576);
  conv(h2,   w3p, b3, h3, nullptr, 128, 7, 256, 6, 2, 0, 1, 1152);
  conv(h3,   w4p, b4, zt, nullptr, 256, 6, 512, 5, 2, 0, 0, 2304);

  // ---- quantize ----
  k_transpose_z<<<grid1((long)16384 * 512), 256, 0, stream>>>(
      zt, zf, 16384 * 512);
  k_norms_bf16<<<grid1(16384), 256, 0, stream>>>(zf, z2, 16384, 512);
  k_vq_argmin<<<dim3(128), 256, 0, stream>>>(zf, cb_bf, z2, c2, ind, 8192, 512);
  k_gather_zq<<<grid1((long)16384 * 512), 256, 0, stream>>>(
      cbf, ind, zq, 16384 * 512);

  // ---- decoder ----
  conv(zq, d1p, d1b, g1, nullptr, 512, 5, 256, 6, 2, 1, 1, 4608);
  conv(g1, d2p, d2b, g2, nullptr, 256, 6, 128, 7, 2, 1, 1, 2304);
  conv(g2, d3p, d3b, g3, nullptr, 128, 7, 64,  8, 2, 1, 1, 1152);
  conv(g3, wop, bo, nullptr, (float*)d_out,
       64, 8, 3, 8, 1, 0, 2, 576);

  // ---- indices appended after x_recon ----
  int* idx_out = (int*)d_out + (size_t)Nb * 3 * 256 * 256;
  k_copy_idx<<<grid1(16384), 256, 0, stream>>>(ind, idx_out, 16384);
}
